// CausalAttention_35545149342293
// MI455X (gfx1250) — matrix operand order repaired and checked by reading, not compiled, not run on hardware
//
#include <hip/hip_runtime.h>
#include <hip/hip_bf16.h>

typedef _Float16 v16h __attribute__((ext_vector_type(16)));
typedef _Float16 v8h  __attribute__((ext_vector_type(8)));
typedef float    v8f  __attribute__((ext_vector_type(8)));

#define WMMA16(a, b, c) \
  __builtin_amdgcn_wmma_f32_16x16x32_f16(false, (a), false, (b), (short)0, (c), false, false)

static __device__ __forceinline__ v16h cat8(v8h lo, v8h hi) {
  return __builtin_shufflevector(lo, hi, 0, 1, 2, 3, 4, 5, 6, 7, 8, 9, 10, 11, 12, 13, 14, 15);
}

// ---------------------------------------------------------------------------
// Kernel 0: one-shot f32 -> f16 conversion (8 elements/thread, vectorized).
// ---------------------------------------------------------------------------
__global__ __launch_bounds__(256) void cvt_f32_to_f16_kernel(
    const float* __restrict__ src, _Float16* __restrict__ dst, int n8) {
  const int i = blockIdx.x * blockDim.x + threadIdx.x;
  if (i < n8) {
    v8f f = *(const v8f*)(src + (size_t)i * 8);
    *(v8h*)(dst + (size_t)i * 8) = __builtin_convertvector(f, v8h);
  }
}

// ---------------------------------------------------------------------------
// Kernel 1: fused QKV projection, y = x @ W^T in f16 WMMA / f32 accumulate.
// blockIdx.z selects Q / K / V.  Q,K stored row-major f16; V stored
// TRANSPOSED (Vt[b][col][s]) so attention's P@V B-frags are contiguous.
//   M = B*S = 8192, N = 1024, K = 1024
//   block tile 128(M) x 128(N), 8 waves as 4(M) x 2(N), wave tile 32x64.
// Inner loop: 8 fragment loads + 8 WMMAs, no conversions.
// ---------------------------------------------------------------------------
__global__ __launch_bounds__(256, 1) void qkv_proj_kernel(
    const _Float16* __restrict__ xh,
    const _Float16* __restrict__ Wqh, const _Float16* __restrict__ Wkh,
    const _Float16* __restrict__ Wvh,
    _Float16* __restrict__ Qh, _Float16* __restrict__ Kh,
    _Float16* __restrict__ Vth) {
  const int D    = 1024;
  const int tid  = threadIdx.x;
  const int w    = tid >> 5;
  const int lane = tid & 31;
  const int ln   = lane & 15;
  const int h    = lane >> 4;
  const int wm   = w >> 1;   // 0..3
  const int wn   = w & 1;    // 0..1
  const int bm0  = blockIdx.x * 128 + wm * 32;
  const int bn0  = blockIdx.y * 128 + wn * 64;
  const _Float16* W = (blockIdx.z == 0) ? Wqh : (blockIdx.z == 1) ? Wkh : Wvh;

  v8f c[2][4];
#pragma unroll
  for (int mt = 0; mt < 2; ++mt)
#pragma unroll
    for (int nt = 0; nt < 4; ++nt) c[mt][nt] = (v8f)(0.0f);

  const _Float16* arow0 = xh + (size_t)(bm0 + ln) * D;
  const _Float16* arow1 = xh + (size_t)(bm0 + 16 + ln) * D;

  for (int kk = 0; kk < D; kk += 32) {
    v16h a[2];
    a[0] = cat8(*(const v8h*)(arow0 + kk + h * 16),
                *(const v8h*)(arow0 + kk + h * 16 + 8));
    a[1] = cat8(*(const v8h*)(arow1 + kk + h * 16),
                *(const v8h*)(arow1 + kk + h * 16 + 8));
    v16h bfr[4];
#pragma unroll
    for (int nt = 0; nt < 4; ++nt)
      bfr[nt] = *(const v16h*)(W + (size_t)(bn0 + nt * 16 + ln) * D + kk + h * 16);
#pragma unroll
    for (int mt = 0; mt < 2; ++mt)
#pragma unroll
      for (int nt = 0; nt < 4; ++nt) c[mt][nt] = WMMA16(a[mt], bfr[nt], c[mt][nt]);
  }

  if (blockIdx.z < 2) {
    _Float16* O = (blockIdx.z == 0) ? Qh : Kh;
#pragma unroll
    for (int mt = 0; mt < 2; ++mt)
#pragma unroll
      for (int nt = 0; nt < 4; ++nt)
#pragma unroll
        for (int v = 0; v < 8; ++v) {
          const int m = bm0 + mt * 16 + v + 8 * h;  // C layout: row = v + 8h
          const int n = bn0 + nt * 16 + ln;
          O[(size_t)m * D + n] = (_Float16)c[mt][nt][v];
        }
  } else {
#pragma unroll
    for (int mt = 0; mt < 2; ++mt)
#pragma unroll
      for (int nt = 0; nt < 4; ++nt) {
        const int m     = bm0 + mt * 16 + 8 * h;  // token of element v=0
        const int batch = m >> 11;                // 2048 tokens per batch
        const int s     = m & 2047;
        const int col   = bn0 + nt * 16 + ln;
        v8h pv = __builtin_convertvector(c[mt][nt], v8h);
        *(v8h*)(Vth + ((size_t)batch * 1024 + col) * 2048 + s) = pv;
      }
  }
}

// ---------------------------------------------------------------------------
// Kernel 2: causal flash attention over d = 1024 (single head), 64-key tiles.
// One block = 16 query rows x all 1024 out cols.  8 waves: each wave owns a
// 128-col output chunk (8 f32 accumulator tiles) and a 128-wide d-slice for
// the score reduction.  Per 64-key tile: 16 partial Q.K^T WMMAs per wave,
// ds_add_f32 reduction in LDS, online softmax (consistent per wave, with a
// uniform no-mask fast path for interior tiles), P routed through
// wave-private LDS for C->A layout, then 16 P@V WMMAs against contiguous Vt.
// ---------------------------------------------------------------------------
__global__ __launch_bounds__(256, 1) void causal_attn_kernel(
    const _Float16* __restrict__ Qh, const _Float16* __restrict__ Kh,
    const _Float16* __restrict__ Vth, float* __restrict__ out) {
  const int D = 1024, S = 2048;
  __shared__ float s_sc[16 * 64];                                  // 4 KB
  __shared__ __align__(32) _Float16 s_p[8][16 * 64];               // 16 KB

  const int tid  = threadIdx.x;
  const int w    = tid >> 5;
  const int lane = tid & 31;
  const int ln   = lane & 15;
  const int h    = lane >> 4;
  const int batch = blockIdx.x >> 7;
  const int q0    = (blockIdx.x & 127) * 16;

  const _Float16* Qb = Qh + ((size_t)batch * S + q0) * D;
  const _Float16* Kb = Kh + (size_t)batch * S * D;
  const _Float16* Vb = Vth + (size_t)batch * D * S;

  // Loop-invariant Q fragments: 16 rows x this wave's 128-wide d-slice.
  v16h aq[4];
#pragma unroll
  for (int dw = 0; dw < 4; ++dw) {
    const _Float16* ab = Qb + (size_t)ln * D + w * 128 + dw * 32;
    aq[dw] = cat8(*(const v8h*)(ab + h * 16), *(const v8h*)(ab + h * 16 + 8));
  }

  v8f acc[8];
#pragma unroll
  for (int t = 0; t < 8; ++t) acc[t] = (v8f)(0.0f);
  float mrow[8], lrow[8];
#pragma unroll
  for (int v = 0; v < 8; ++v) { mrow[v] = -1e30f; lrow[v] = 0.0f; }

  const int nkt = (q0 + 16 + 63) >> 6;  // 64-key tiles; skip fully masked
  for (int kt = 0; kt < nkt; ++kt) {
    const int k0 = kt * 64;
    const bool nomask = (k0 + 63 <= q0);  // uniform: whole tile unmasked

    // Prefetch next key tile's K rows for this wave's d-slice.
    if (kt + 1 < nkt) {
      __builtin_prefetch(Kb + (size_t)(k0 + 64 + lane) * D + w * 128, 0, 1);
      __builtin_prefetch(Kb + (size_t)(k0 + 96 + lane) * D + w * 128, 0, 1);
    }

#pragma unroll
    for (int i = 0; i < 4; ++i) s_sc[tid + i * 256] = 0.0f;
    __syncthreads();

    // Partial scores over this wave's d-slice [w*128, w*128+128).
    v8f sc[4];
#pragma unroll
    for (int t = 0; t < 4; ++t) sc[t] = (v8f)(0.0f);
#pragma unroll
    for (int dw = 0; dw < 4; ++dw) {
      const int d0 = w * 128 + dw * 32;
#pragma unroll
      for (int t = 0; t < 4; ++t) {
        const _Float16* kb = Kb + (size_t)(k0 + t * 16 + ln) * D + d0 + h * 16;
        sc[t] = WMMA16(aq[dw], *(const v16h*)kb, sc[t]);
      }
    }
#pragma unroll
    for (int t = 0; t < 4; ++t)
#pragma unroll
      for (int v = 0; v < 8; ++v)
        atomicAdd(&s_sc[(v + 8 * h) * 64 + t * 16 + ln], sc[t][v]);
    __syncthreads();

    // Online softmax on the complete 16x64 score tile (C layout reads).
#pragma unroll
    for (int v = 0; v < 8; ++v) {
      const int mg = q0 + v + 8 * h;
      const int rb = (v + 8 * h) * 64;
      float sv[4];
#pragma unroll
      for (int t = 0; t < 4; ++t) {
        float s = s_sc[rb + t * 16 + ln] * 0.03125f;       // 1/sqrt(1024)
        if (!nomask && (k0 + t * 16 + ln > mg)) s = -1e30f;
        sv[t] = s;
      }
      float mx = fmaxf(fmaxf(sv[0], sv[1]), fmaxf(sv[2], sv[3]));
#pragma unroll
      for (int off = 1; off < 16; off <<= 1) mx = fmaxf(mx, __shfl_xor(mx, off));
      const float nm = fmaxf(mrow[v], mx);
      const float f  = __expf(mrow[v] - nm);
      float p[4], rs = 0.0f;
#pragma unroll
      for (int t = 0; t < 4; ++t) { p[t] = __expf(sv[t] - nm); rs += p[t]; }
#pragma unroll
      for (int off = 1; off < 16; off <<= 1) rs += __shfl_xor(rs, off);
      lrow[v] = lrow[v] * f + rs;
      mrow[v] = nm;
#pragma unroll
      for (int t2 = 0; t2 < 8; ++t2) acc[t2][v] *= f;
#pragma unroll
      for (int t = 0; t < 4; ++t)
        s_p[w][rb + t * 16 + ln] = (_Float16)p[t];
    }

    // Read P back as two 16x32 A fragments (wave-local LDS, DScnt-ordered).
    v16h pa[2];
#pragma unroll
    for (int kc = 0; kc < 2; ++kc)
      pa[kc] = cat8(*(const v8h*)&s_p[w][ln * 64 + kc * 32 + h * 16],
                    *(const v8h*)&s_p[w][ln * 64 + kc * 32 + h * 16 + 8]);

    // P @ V for this wave's 128-col chunk; Vt rows are key-contiguous.
#pragma unroll
    for (int t2 = 0; t2 < 8; ++t2) {
      const _Float16* vb = Vb + (size_t)(w * 128 + t2 * 16 + ln) * S + k0;
      acc[t2] = WMMA16(pa[0], *(const v16h*)(vb + h * 16), acc[t2]);
      acc[t2] = WMMA16(pa[1], *(const v16h*)(vb + 32 + h * 16), acc[t2]);
    }
    __syncthreads();  // protect s_sc re-zero of next iteration
  }

#pragma unroll
  for (int t2 = 0; t2 < 8; ++t2)
#pragma unroll
    for (int v = 0; v < 8; ++v) {
      const size_t row = (size_t)batch * S + q0 + v + 8 * h;
      out[row * D + w * 128 + t2 * 16 + ln] = acc[t2][v] / lrow[v];
    }
}

extern "C" void kernel_launch(void* const* d_in, const int* in_sizes, int n_in,
                              void* d_out, int out_size, void* d_ws, size_t ws_size,
                              hipStream_t stream) {
  (void)in_sizes; (void)n_in; (void)out_size; (void)ws_size;
  const float* x  = (const float*)d_in[0];
  const float* Wq = (const float*)d_in[1];
  const float* Wk = (const float*)d_in[2];
  const float* Wv = (const float*)d_in[3];
  float* out = (float*)d_out;

  // Workspace layout (f16): Qh/Kh/Vth 8Mi each, xh 8Mi, W*h 1Mi each = 70 MB.
  const size_t elems = (size_t)4 * 2048 * 1024;  // 8Mi
  const size_t welem = (size_t)1024 * 1024;      // 1Mi
  _Float16* Qh  = (_Float16*)d_ws;
  _Float16* Kh  = Qh + elems;
  _Float16* Vth = Kh + elems;
  _Float16* xhp = Vth + elems;
  _Float16* Wqh = xhp + elems;
  _Float16* Wkh = Wqh + welem;
  _Float16* Wvh = Wkh + welem;

  cvt_f32_to_f16_kernel<<<(int)(elems / 8 / 256), 256, 0, stream>>>(x, xhp, (int)(elems / 8));
  cvt_f32_to_f16_kernel<<<(int)(welem / 8 / 256), 256, 0, stream>>>(Wq, Wqh, (int)(welem / 8));
  cvt_f32_to_f16_kernel<<<(int)(welem / 8 / 256), 256, 0, stream>>>(Wk, Wkh, (int)(welem / 8));
  cvt_f32_to_f16_kernel<<<(int)(welem / 8 / 256), 256, 0, stream>>>(Wv, Wvh, (int)(welem / 8));

  // QKV projection: M=8192 -> 64 blocks of 128; N=1024 -> 8 blocks; z = Q/K/V
  dim3 grid1(64, 8, 3);
  qkv_proj_kernel<<<grid1, 256, 0, stream>>>(xhp, Wqh, Wkh, Wvh, Qh, Kh, Vth);

  // Attention: one block per 16 query rows: 4 * (2048/16) = 512 blocks.
  dim3 grid2(512);
  causal_attn_kernel<<<grid2, 256, 0, stream>>>(Qh, Kh, Vth, out);
}
